// LagrangianODESolver_28870770164322
// MI455X (gfx1250) — matrix operand order repaired and checked by reading, not compiled, not run on hardware
//
#include <hip/hip_runtime.h>
#include <hip/hip_bf16.h>

// LagrangianODESolver for MI455X (gfx1250, wave32, WMMA + TDM).
// All GEMMs use v_wmma_f32_16x16x32_bf16 (bf16 in, f32 accumulate).

#define B_N   1024
#define T_N   64
#define D_N   512
#define HID_N 256
#define NH_N  4
#define HD_N  128
#define DT    0.05f   // t_steps[1]-t_steps[0] in the reference

typedef unsigned short u16;
typedef unsigned int   u32;
typedef __attribute__((ext_vector_type(16))) __bf16 bf16x16;
typedef __attribute__((ext_vector_type(8)))  float  f32x8;
typedef __attribute__((ext_vector_type(4)))  u32    u32x4;   // trivial 16B vector
typedef __attribute__((ext_vector_type(4)))  u32    uv4;     // TDM D# group 0
typedef __attribute__((ext_vector_type(8)))  int    iv8;     // TDM D# group 1 / arg5
typedef __attribute__((ext_vector_type(4)))  int    iv4;     // TDM D# groups 2/3

#if defined(__has_builtin)
#if __has_builtin(__builtin_amdgcn_tensor_load_to_lds) && \
    __has_builtin(__builtin_amdgcn_s_wait_tensorcnt)
#define HAVE_TDM 1
#endif
#endif

union Frag {            // 16 bf16 = 32 bytes = 2 x 16B chunks
  bf16x16 v;
  u32x4   q[2];
};

__device__ inline u16 f2bf(float x) {            // f32 -> bf16 RNE
  u32 u = __float_as_uint(x);
  u = (u + 0x7FFFu + ((u >> 16) & 1u)) >> 16;
  return (u16)u;
}

__device__ inline float gelu_f(float x) {        // exact erf GELU
  return 0.5f * x * (1.0f + erff(x * 0.70710678118654752f));
}

// A fragment: 16x32 bf16 tile from row-major source (row stride ld elems).
// Lane L holds row M=L&15; lanes 0-15 get K halves {0..7,16..23}, lanes 16-31 {8..15,24..31}.
__device__ inline bf16x16 load_a(const u16* base, int ld, int kbase, int lane) {
  int m = lane & 15, kh = lane >> 4;
  const u16* p = base + m * ld + kbase + kh * 8;
  Frag f;
  f.q[0] = *(const u32x4*)(p);        // K = kbase + kh*8 + 0..7
  f.q[1] = *(const u32x4*)(p + 16);   // K = kbase + 16 + kh*8 + 0..7
  return f.v;
}

// B fragment: 32x16 bf16 where column n of B = row (nbase+n) of source
// (source row-major [N][K], row stride ld). Lanes 0-15: K {0..7,16..23}, lanes 16-31: {8..15,24..31}.
__device__ inline bf16x16 load_b(const u16* base, int ld, int nbase, int kbase, int lane) {
  int n = lane & 15, kh = lane >> 4;
  const u16* p = base + (size_t)(nbase + n) * ld + kbase + kh * 8;
  Frag f;
  f.q[0] = *(const u32x4*)(p);
  f.q[1] = *(const u32x4*)(p + 16);
  return f.v;
}

__device__ inline f32x8 wmma_bf16(bf16x16 a, bf16x16 b, f32x8 c) {
  return __builtin_amdgcn_wmma_f32_16x16x32_bf16(false, a, false, b, (short)0, c,
                                                 false, false);
}

// ---------------------------------------------------------------- weights cvt
__global__ __launch_bounds__(256) void cvt_kernel(const float* __restrict__ src,
                                                  u16* __restrict__ dst, int n) {
  int i = blockIdx.x * 256 + threadIdx.x;
  if (i < n) dst[i] = f2bf(src[i]);
}

// ---------------------------------------------------------------- ODE / RK4
// One workgroup (8 wave32) owns 16 batch rows and runs all 64 RK4 steps fused.
// State z_t / RK accumulator / running sum live in registers, activations
// staged in LDS as bf16 WMMA A tiles, W1/W3 streamed from L2 as B fragments,
// W2 (128 KB bf16) parked in LDS via the Tensor Data Mover (read 256x per WG).
__global__ __launch_bounds__(256) void ode_kernel(
    const float* __restrict__ z_init,
    const u16* __restrict__ W1, const float* __restrict__ b1,
    const u16* __restrict__ W2, const float* __restrict__ b2,
    const u16* __restrict__ W3, const float* __restrict__ b3,
    float* __restrict__ z_stack, float* __restrict__ z_sum) {
  __shared__ float sF[16 * D_N];                       // field output (f32)
  __shared__ __align__(16) u16 sZe[16 * (D_N + 8)];    // z_eval (bf16 A tile)
  __shared__ __align__(16) u16 sH1[16 * (HID_N + 8)];
  __shared__ __align__(16) u16 sH2[16 * (HID_N + 8)];
  __shared__ __align__(16) u16 sW2[HID_N * HID_N];     // LDS-resident W2 (bf16)
  __shared__ float sP1[16 * 16];
  __shared__ float sP2[16 * 16];
  __shared__ float sScale[16];

  const int tid  = threadIdx.x;
  const int wave = tid >> 5, lane = tid & 31;
  const int rbase = blockIdx.x * 16;

#ifdef HAVE_TDM
  // ---- TDM: async-copy W2 [256x256 bf16] global -> LDS (D# per ISA ch.8)
  if (wave == 0) {
    u32 lds_off = (u32)(size_t)(void*)sW2;           // flat[31:0] == LDS offset
    unsigned long long ga = (unsigned long long)(size_t)W2;
    uv4 g0 = { 1u,                                   // count=1, user descriptor
               lds_off,                               // lds_addr
               (u32)ga,                               // global_addr[31:0]
               (u32)((ga >> 32) & 0x1FFFFFFu) | (2u << 30) };  // addr[56:32]|type=2
    iv8 g1 = { (int)(1u << 16),          // data_size=1 (2 bytes/elem)
               (int)((u32)HID_N << 16),  // tensor_dim0 = 256   (bits 79:48 lo)
               (int)((u32)HID_N << 16),  // tensor_dim1 = 256   (bits 111:80 lo)
               (int)((u32)HID_N << 16),  // tile_dim0   = 256   (bits 127:112)
               (int)HID_N,               // tile_dim1   = 256, tile_dim2 = 0
               (int)HID_N,               // tensor_dim0_stride = 256
               0, 0 };                   // stride hi / tensor_dim1_stride = 0
    iv4 gz4 = { 0, 0, 0, 0 };            // groups 2/3 unused (2D tensor)
    iv8 gz8 = { 0, 0, 0, 0, 0, 0, 0, 0 };
    __builtin_amdgcn_tensor_load_to_lds(g0, g1, gz4, gz4, gz8, 0);
    __builtin_amdgcn_s_wait_tensorcnt((short)0);
  }
#else
  // Fallback: cooperative vectorized copy of W2 into LDS.
  for (int i = tid; i < (HID_N * HID_N) / 8; i += 256)
    *(u32x4*)(sW2 + i * 8) = *(const u32x4*)(W2 + i * 8);
#endif

  // ---- input-adaptive scale: clip(norm / (var_unbiased + 1e-6), 0.8, 1.2)
  {
    int row = tid >> 4, sub = tid & 15;
    float s1 = 0.f, s2 = 0.f;
    const float* zr = z_init + (size_t)(rbase + row) * D_N + sub * 32;
    for (int j = 0; j < 32; ++j) { float v = zr[j]; s1 += v; s2 += v * v; }
    sP1[row * 16 + sub] = s1;
    sP2[row * 16 + sub] = s2;
  }
  __syncthreads();   // also publishes sW2 to all waves
  if (tid < 16) {
    float s1 = 0.f, s2 = 0.f;
    for (int j = 0; j < 16; ++j) { s1 += sP1[tid * 16 + j]; s2 += sP2[tid * 16 + j]; }
    float mean = s1 / (float)D_N;
    float nrm  = sqrtf(s2);
    float var  = (s2 - (float)D_N * mean * mean) / (float)(D_N - 1);
    float sc   = nrm / (var + 1e-6f);
    sScale[tid] = fminf(fmaxf(sc, 0.8f), 1.2f);
  }
  __syncthreads();

  float regZt[32], regAcc[32], regSum[32];
#pragma unroll
  for (int j = 0; j < 32; ++j) {
    int i = tid + 256 * j;
    int r = i >> 9, d = i & (D_N - 1);
    float v = z_init[(size_t)(rbase + r) * D_N + d] * sScale[r];
    regZt[j]  = v;
    regSum[j] = 0.f;
    regAcc[j] = 0.f;
    sZe[r * (D_N + 8) + d] = f2bf(v);
  }
  __syncthreads();

  // f(z) = W3 * gelu(W2 * gelu(W1 z + b1) + b2) + b3, result -> sF (f32)
  auto field = [&]() {
    for (int nt = wave; nt < HID_N / 16; nt += 8) {        // h1: [16 x 256]
      f32x8 acc = {};
      for (int kk = 0; kk < D_N; kk += 32) {
        if (kk + 32 < D_N)   // warm next B block (global_prefetch_b8)
          __builtin_prefetch(W1 + (size_t)(nt * 16 + (lane & 15)) * D_N + kk + 32, 0, 0);
        acc = wmma_bf16(load_a(sZe, D_N + 8, kk, lane),
                        load_b(W1, D_N, nt * 16, kk, lane), acc);
      }
      int n = nt * 16 + (lane & 15), kh = lane >> 4;
      float bias = b1[n];
#pragma unroll
      for (int r = 0; r < 8; ++r)
        sH1[(r + 8 * kh) * (HID_N + 8) + n] = f2bf(gelu_f(acc[r] + bias));
    }
    __syncthreads();
    for (int nt = wave; nt < HID_N / 16; nt += 8) {        // h2: [16 x 256], W2 in LDS
      f32x8 acc = {};
      for (int kk = 0; kk < HID_N; kk += 32)
        acc = wmma_bf16(load_a(sH1, HID_N + 8, kk, lane),
                        load_b(sW2, HID_N, nt * 16, kk, lane), acc);
      int n = nt * 16 + (lane & 15), kh = lane >> 4;
      float bias = b2[n];
#pragma unroll
      for (int r = 0; r < 8; ++r)
        sH2[(r + 8 * kh) * (HID_N + 8) + n] = f2bf(gelu_f(acc[r] + bias));
    }
    __syncthreads();
    for (int nt = wave; nt < D_N / 16; nt += 8) {          // f: [16 x 512]
      f32x8 acc = {};
      for (int kk = 0; kk < HID_N; kk += 32)
        acc = wmma_bf16(load_a(sH2, HID_N + 8, kk, lane),
                        load_b(W3, HID_N, nt * 16, kk, lane), acc);
      int n = nt * 16 + (lane & 15), kh = lane >> 4;
      float bias = b3[n];
#pragma unroll
      for (int r = 0; r < 8; ++r)
        sF[(r + 8 * kh) * D_N + n] = acc[r] + bias;
    }
    __syncthreads();
  };

  // after a field eval: acc (+)= wsum*k ; z_eval = z_t + ceval*k
  auto stage = [&](float wsum, float ceval, bool first) {
#pragma unroll
    for (int j = 0; j < 32; ++j) {
      int i = tid + 256 * j;
      int r = i >> 9, d = i & (D_N - 1);
      float k = sF[i];
      regAcc[j] = (first ? regZt[j] : regAcc[j]) + wsum * k;
      sZe[r * (D_N + 8) + d] = f2bf(regZt[j] + ceval * k);
    }
    __syncthreads();
  };

  const float dt6 = DT / 6.0f, dt3 = DT / 3.0f, dth = DT * 0.5f;
  for (int t = 0; t < T_N; ++t) {
    field();                  // k1
    stage(dt6, dth, true);
    field();                  // k2
    stage(dt3, dth, false);
    field();                  // k3
    stage(dt3, DT, false);
    field();                  // k4
#pragma unroll
    for (int j = 0; j < 32; ++j) {
      int i = tid + 256 * j;
      int r = i >> 9, d = i & (D_N - 1);
      float zn = regAcc[j] + dt6 * sF[i];
      zn = (zn != zn) ? 0.f : zn;            // nan -> 0
      regZt[j]  = zn;
      regSum[j] += zn;
      sZe[r * (D_N + 8) + d] = f2bf(zn);     // z_eval for next k1
      z_stack[((size_t)(rbase + r) * T_N + t) * D_N + d] = zn;
    }
    __syncthreads();
  }
#pragma unroll
  for (int j = 0; j < 32; ++j) {
    int i = tid + 256 * j;
    int r = i >> 9, d = i & (D_N - 1);
    z_sum[(size_t)(rbase + r) * D_N + d] = regSum[j];
  }
}

// ---------------------------------------------------------------- z_dc
__global__ __launch_bounds__(256) void zdc_kernel(
    float* __restrict__ z_stack, const float* __restrict__ z_sum,
    const float* __restrict__ z_init, u16* __restrict__ z_dc_h) {
  int idx = blockIdx.x * 256 + threadIdx.x;
  int b = idx / (T_N * D_N);
  int d = idx & (D_N - 1);
  float mean = z_sum[(size_t)b * D_N + d] * (1.0f / (float)T_N);
  float v = z_stack[idx] - 0.05f * (mean - z_init[(size_t)b * D_N + d]);
  z_stack[idx] = v;            // becomes z_dc (f32), consumed by out_kernel
  z_dc_h[idx]  = f2bf(v);      // bf16 copy for qkv GEMM
}

// ---------------------------------------------------------------- QKV GEMM
// [65536,512] x [512,1536] ; grid (M/16, 96/8), one N tile per wave.
__global__ __launch_bounds__(256) void qkv_kernel(
    const u16* __restrict__ A, const u16* __restrict__ Wqkv,
    const float* __restrict__ bqkv, u16* __restrict__ qkv) {
  const int lane = threadIdx.x & 31, wave = threadIdx.x >> 5;
  const int mbase = blockIdx.x * 16;
  const int nt = blockIdx.y * 8 + wave;
  const u16* Ab = A + (size_t)mbase * D_N;
  f32x8 acc = {};
  for (int kk = 0; kk < D_N; kk += 32) {
    if (kk + 32 < D_N)
      __builtin_prefetch(Wqkv + (size_t)(nt * 16 + (lane & 15)) * D_N + kk + 32, 0, 0);
    acc = wmma_bf16(load_a(Ab, D_N, kk, lane),
                    load_b(Wqkv, D_N, nt * 16, kk, lane), acc);
  }
  int n = nt * 16 + (lane & 15), kh = lane >> 4;
  float bias = bqkv[n];
#pragma unroll
  for (int r = 0; r < 8; ++r)
    qkv[(size_t)(mbase + r + 8 * kh) * (3 * D_N) + n] = f2bf(acc[r] + bias);
}

// ---------------------------------------------------------------- attention
// One workgroup per (b, h): S = QK^T/sqrt(HD) via WMMA, LDS softmax, ctx = S V.
__global__ __launch_bounds__(256) void attn_kernel(
    const u16* __restrict__ qkv, u16* __restrict__ ctx) {
  __shared__ __align__(16) u16 sQ[T_N * (HD_N + 8)];
  __shared__ __align__(16) u16 sK[T_N * (HD_N + 8)];
  __shared__ __align__(16) u16 sVt[HD_N * (T_N + 8)];   // V transposed [d][t]
  __shared__ float sS[T_N * T_N];
  __shared__ __align__(16) u16 sSb[T_N * (T_N + 8)];
  __shared__ float sR[T_N * 4];

  const int tid = threadIdx.x;
  const int lane = tid & 31, wave = tid >> 5;
  const int b = blockIdx.x, h = blockIdx.y;
  const size_t rs = 3 * D_N;
  const u16* baseq = qkv + (size_t)b * T_N * rs + h * HD_N;

  for (int i = tid; i < T_N * (HD_N / 8); i += 256) {
    int t = i >> 4, c = i & 15;
    const u16* src = baseq + (size_t)t * rs + c * 8;
    *(u32x4*)(sQ + t * (HD_N + 8) + c * 8) = *(const u32x4*)(src);
    *(u32x4*)(sK + t * (HD_N + 8) + c * 8) = *(const u32x4*)(src + D_N);
    union { u32x4 q; u16 s[8]; } tmp;
    tmp.q = *(const u32x4*)(src + 2 * D_N);
#pragma unroll
    for (int j = 0; j < 8; ++j) sVt[(c * 8 + j) * (T_N + 8) + t] = tmp.s[j];
  }
  __syncthreads();

  for (int id = wave; id < 16; id += 8) {               // S = Q K^T * scale
    int mt = id >> 2, nt = id & 3;
    f32x8 acc = {};
    for (int kk = 0; kk < HD_N; kk += 32)
      acc = wmma_bf16(load_a(sQ + mt * 16 * (HD_N + 8), HD_N + 8, kk, lane),
                      load_b(sK, HD_N + 8, nt * 16, kk, lane), acc);
    int n = nt * 16 + (lane & 15), kh = lane >> 4;
#pragma unroll
    for (int r = 0; r < 8; ++r)
      sS[(mt * 16 + r + 8 * kh) * T_N + n] = acc[r] * 0.088388347648318447f;
  }
  __syncthreads();

  {                                                     // row softmax
    int row = tid >> 2, sub = tid & 3;
    float mx = -3.4e38f;
    for (int j = 0; j < 16; ++j) mx = fmaxf(mx, sS[row * T_N + sub * 16 + j]);
    sR[row * 4 + sub] = mx;
    __syncthreads();
    float rm = fmaxf(fmaxf(sR[row * 4 + 0], sR[row * 4 + 1]),
                     fmaxf(sR[row * 4 + 2], sR[row * 4 + 3]));
    __syncthreads();
    float sm = 0.f;
    for (int j = 0; j < 16; ++j) {
      float e = __expf(sS[row * T_N + sub * 16 + j] - rm);
      sS[row * T_N + sub * 16 + j] = e;
      sm += e;
    }
    sR[row * 4 + sub] = sm;
    __syncthreads();
    float inv = 1.0f / (sR[row * 4 + 0] + sR[row * 4 + 1] +
                        sR[row * 4 + 2] + sR[row * 4 + 3]);
    for (int j = 0; j < 16; ++j)
      sSb[row * (T_N + 8) + sub * 16 + j] = f2bf(sS[row * T_N + sub * 16 + j] * inv);
  }
  __syncthreads();

  for (int id = wave; id < 32; id += 8) {               // ctx = att V  [64 x 128]
    int mt = id >> 3, nt = id & 7;
    f32x8 acc = {};
    for (int kk = 0; kk < T_N; kk += 32)
      acc = wmma_bf16(load_a(sSb + mt * 16 * (T_N + 8), T_N + 8, kk, lane),
                      load_b(sVt, T_N + 8, nt * 16, kk, lane), acc);
    int n = nt * 16 + (lane & 15), kh = lane >> 4;
#pragma unroll
    for (int r = 0; r < 8; ++r) {
      int t = mt * 16 + r + 8 * kh;
      ctx[((size_t)(b * T_N + t)) * D_N + h * HD_N + nt * 16 + n] = f2bf(acc[r]);
    }
  }
}

// ------------------------------------------- out-proj + residual + LayerNorm
__global__ __launch_bounds__(256) void out_kernel(
    const u16* __restrict__ ctx, const u16* __restrict__ Wo,
    const float* __restrict__ bo, const float* __restrict__ z_dc,
    const float* __restrict__ gamma, const float* __restrict__ beta,
    float* __restrict__ out) {
  __shared__ float sX[16 * D_N];
  __shared__ float sP1[16 * 16];
  __shared__ float sP2[16 * 16];
  const int tid = threadIdx.x;
  const int lane = tid & 31, wave = tid >> 5;
  const int mbase = blockIdx.x * 16;
  const u16* Ab = ctx + (size_t)mbase * D_N;

  for (int nt = wave; nt < D_N / 16; nt += 8) {
    f32x8 acc = {};
    for (int kk = 0; kk < D_N; kk += 32)
      acc = wmma_bf16(load_a(Ab, D_N, kk, lane),
                      load_b(Wo, D_N, nt * 16, kk, lane), acc);
    int n = nt * 16 + (lane & 15), kh = lane >> 4;
    float bias = bo[n];
#pragma unroll
    for (int r = 0; r < 8; ++r) {
      int m = r + 8 * kh;
      sX[m * D_N + n] = acc[r] + bias + z_dc[(size_t)(mbase + m) * D_N + n];
    }
  }
  __syncthreads();

  int row = tid >> 4, sub = tid & 15;
  float s1 = 0.f, s2 = 0.f;
  for (int j = 0; j < 32; ++j) {
    float v = sX[row * D_N + sub * 32 + j];
    s1 += v; s2 += v * v;
  }
  sP1[row * 16 + sub] = s1;
  sP2[row * 16 + sub] = s2;
  __syncthreads();
  float t1 = 0.f, t2 = 0.f;
  for (int j = 0; j < 16; ++j) { t1 += sP1[row * 16 + j]; t2 += sP2[row * 16 + j]; }
  float mean = t1 * (1.0f / (float)D_N);
  float var  = t2 * (1.0f / (float)D_N) - mean * mean;   // biased, like jnp.var
  float rstd = rsqrtf(var + 1e-5f);
  for (int j = 0; j < 32; ++j) {
    int d = sub * 32 + j;
    float v = (sX[row * D_N + d] - mean) * rstd * gamma[d] + beta[d];
    v = (v != v) ? 0.f : v;
    out[(size_t)(mbase + row) * D_N + d] = v;
  }
}

// ---------------------------------------------------------------- launcher
extern "C" void kernel_launch(void* const* d_in, const int* in_sizes, int n_in,
                              void* d_out, int out_size, void* d_ws, size_t ws_size,
                              hipStream_t stream) {
  (void)in_sizes; (void)n_in; (void)out_size; (void)ws_size;
  const float* z_init = (const float*)d_in[0];
  // d_in[1] = t_steps (dt = 0.05 hardcoded to match reference arange * 0.05)
  const float* W1f   = (const float*)d_in[2];
  const float* b1    = (const float*)d_in[3];
  const float* W2f   = (const float*)d_in[4];
  const float* b2    = (const float*)d_in[5];
  const float* W3f   = (const float*)d_in[6];
  const float* b3    = (const float*)d_in[7];
  const float* Wqkvf = (const float*)d_in[8];
  const float* bqkv  = (const float*)d_in[9];
  const float* Wof   = (const float*)d_in[10];
  const float* bo    = (const float*)d_in[11];
  const float* gamma = (const float*)d_in[12];
  const float* beta  = (const float*)d_in[13];

  char* ws = (char*)d_ws;
  size_t off = 0;
  auto alloc = [&](size_t bytes) {
    void* p = ws + off;
    off = (off + bytes + 255) & ~(size_t)255;
    return p;
  };
  u16*   wB1     = (u16*)alloc((size_t)HID_N * D_N * 2);
  u16*   wB2     = (u16*)alloc((size_t)HID_N * HID_N * 2);
  u16*   wB3     = (u16*)alloc((size_t)D_N * HID_N * 2);
  u16*   wBqkv   = (u16*)alloc((size_t)3 * D_N * D_N * 2);
  u16*   wBo     = (u16*)alloc((size_t)D_N * D_N * 2);
  float* z_sum   = (float*)alloc((size_t)B_N * D_N * 4);
  float* z_stack = (float*)alloc((size_t)B_N * T_N * D_N * 4);   // becomes z_dc f32
  u16*   z_dc_h  = (u16*)alloc((size_t)B_N * T_N * D_N * 2);
  u16*   qkvb    = (u16*)alloc((size_t)B_N * T_N * 3 * D_N * 2);
  u16*   ctxb    = (u16*)alloc((size_t)B_N * T_N * D_N * 2);

  cvt_kernel<<<(HID_N * D_N + 255) / 256, 256, 0, stream>>>(W1f, wB1, HID_N * D_N);
  cvt_kernel<<<(HID_N * HID_N + 255) / 256, 256, 0, stream>>>(W2f, wB2, HID_N * HID_N);
  cvt_kernel<<<(D_N * HID_N + 255) / 256, 256, 0, stream>>>(W3f, wB3, D_N * HID_N);
  cvt_kernel<<<(3 * D_N * D_N + 255) / 256, 256, 0, stream>>>(Wqkvf, wBqkv, 3 * D_N * D_N);
  cvt_kernel<<<(D_N * D_N + 255) / 256, 256, 0, stream>>>(Wof, wBo, D_N * D_N);

  ode_kernel<<<B_N / 16, 256, 0, stream>>>(z_init, wB1, b1, wB2, b2, wB3, b3,
                                           z_stack, z_sum);
  zdc_kernel<<<(B_N * T_N * D_N) / 256, 256, 0, stream>>>(z_stack, z_sum, z_init,
                                                          z_dc_h);
  qkv_kernel<<<dim3((B_N * T_N) / 16, (3 * D_N) / 128), 256, 0, stream>>>(
      z_dc_h, wBqkv, bqkv, qkvb);
  attn_kernel<<<dim3(B_N, NH_N), 256, 0, stream>>>(qkvb, ctxb);
  out_kernel<<<(B_N * T_N) / 16, 256, 0, stream>>>(ctxb, wBo, bo, z_stack,
                                                   gamma, beta, (float*)d_out);
}
